// VADetector_14319420965328
// MI455X (gfx1250) — compile-verified, matrix operand order checked by reading, not run on hardware
//
#include <hip/hip_runtime.h>
#include <hip/hip_bf16.h>
#include <stdint.h>

typedef __attribute__((ext_vector_type(16))) _Float16 v16h;
typedef __attribute__((ext_vector_type(8)))  float    v8f;

#define L_TOTAL   1310720
#define C_LEN     512
#define WARM      64
#define TILE      64
#define NTILES    9                       // 1 warm-up tile + 8 emit tiles
#define NCHUNK    (L_TOTAL / C_LEN)       // 2560
#define WAVES_PB  8
#define CHUNKS_PB (2 * WAVES_PB)          // 16 chunks per 256-thread block
#define NBLOCKS   (NCHUNK / CHUNKS_PB)    // 160

// ds_swizzle group-of-32 XOR pattern: offset = (xor<<10) | and_mask(0x1F)
#define SWZ_MIN(m, imm)                                                         \
  (m) = fminf((m), __int_as_float(                                              \
            __builtin_amdgcn_ds_swizzle(__float_as_int(m), (imm))))

__global__ __launch_bounds__(256) void va_viterbi_kernel(
    const float* __restrict__ y, const float* __restrict__ gammap,
    float* __restrict__ out)
{
  const int tid   = threadIdx.x;
  const int lane  = tid & 31;
  const int wv    = tid >> 5;
  const int hi    = (lane >> 4) & 1;      // which chunk of this wave
  const int l15   = lane & 15;            // state index p for this lane
  const int gwave = blockIdx.x * WAVES_PB + wv;
  const int chunk = gwave * 2 + hi;
  const int t0    = chunk * C_LEN;
  const int wstart = (chunk == 0) ? 0 : (t0 - WARM);

  __shared__ float lds_y[WAVES_PB][2][2][TILE];   // 8 KB: wave x chunk x dblbuf

  // ---------------- isi[p] via WMMA (f16 hi/lo split -> ~f32 accuracy) ------
  const float gamma = gammap[0];
  float hf0 = 1.0f;
  float hf1 = __expf(-gamma);
  float hf2 = __expf(-2.0f * gamma);
  float hf3 = __expf(-3.0f * gamma);

  _Float16 z16 = (_Float16)0.0f;
  _Float16 hh0 = (_Float16)hf0, hh1 = (_Float16)hf1,
           hh2 = (_Float16)hf2, hh3 = (_Float16)hf3;
  _Float16 hl0 = (_Float16)(hf0 - (float)hh0);
  _Float16 hl1 = (_Float16)(hf1 - (float)hh1);
  _Float16 hl2 = (_Float16)(hf2 - (float)hh2);
  _Float16 hl3 = (_Float16)(hf3 - (float)hh3);

  const int  M    = l15;
  const bool lo16 = (lane < 16);
  // new_mat[M][m] = 1 - 2*((M >> (3-m)) & 1)
  float nm0 = 1.0f - 2.0f * (float)((M >> 3) & 1);
  float nm1 = 1.0f - 2.0f * (float)((M >> 2) & 1);
  float nm2 = 1.0f - 2.0f * (float)((M >> 1) & 1);
  float nm3 = 1.0f - 2.0f * (float)((M >> 0) & 1);

  v16h A, Bh, Bl;
#pragma unroll
  for (int e = 0; e < 16; ++e) { A[e] = z16; Bh[e] = z16; Bl[e] = z16; }
  // A: 16x32 f16, lanes 0-15 hold K=0..3 in elems 0..3 (M = lane)
  A[0] = lo16 ? (_Float16)nm0 : z16;
  A[1] = lo16 ? (_Float16)nm1 : z16;
  A[2] = lo16 ? (_Float16)nm2 : z16;
  A[3] = lo16 ? (_Float16)nm3 : z16;
  // B: 32x16 f16, lanes 0-15 hold rows K=0..3 (same value for every column N)
  Bh[0] = lo16 ? hh0 : z16;  Bh[1] = lo16 ? hh1 : z16;
  Bh[2] = lo16 ? hh2 : z16;  Bh[3] = lo16 ? hh3 : z16;
  Bl[0] = lo16 ? hl0 : z16;  Bl[1] = lo16 ? hl1 : z16;
  Bl[2] = lo16 ? hl2 : z16;  Bl[3] = lo16 ? hl3 : z16;

  v8f acc = {};
  acc = __builtin_amdgcn_wmma_f32_16x16x32_f16(false, A, false, Bh,
                                               (short)0, acc, false, false);
  acc = __builtin_amdgcn_wmma_f32_16x16x32_f16(false, A, false, Bl,
                                               (short)0, acc, false, false);

  // D layout: vgpr r => lanes 0-15: M=r ; lanes 16-31: M=r+8. Extract isi[l15].
  float locv = 0.0f, remv = 0.0f;
#pragma unroll
  for (int r = 0; r < 8; ++r) {
    float o = __shfl_xor(acc[r], 16);
    locv = ((lane & 7) == r) ? acc[r] : locv;
    remv = ((lane & 7) == r) ? o      : remv;
  }
  const bool isLocal = (((lane >> 3) & 1) == ((lane >> 4) & 1));
  const float isi_p  = isLocal ? locv : remv;

  // ---------------- precomputed ACS gather indices (bytes, ds_bpermute) -----
  const int halfBase = lane & 16;
  const int i0 = (halfBase + (l15 >> 1)) << 2;   // pred0 = p>>1
  const int i1 = i0 + 32;                        // pred1 = (p>>1)+8

  // ---------------- async double-buffered y staging -------------------------
  unsigned ldsBase[2];
  ldsBase[0] = (unsigned)(uintptr_t)(&lds_y[wv][hi][0][0]);
  ldsBase[1] = (unsigned)(uintptr_t)(&lds_y[wv][hi][1][0]);

  auto tile_start = [&](int k) -> int {
    return (k == 0) ? wstart : (t0 + (k - 1) * TILE);
  };
  auto issue = [&](int k) {
    const float* g  = y + tile_start(k) + (l15 << 2);   // 4 floats per lane
    unsigned     lo = ldsBase[k & 1] + ((unsigned)l15 << 4);
    asm volatile("global_load_async_to_lds_b128 %0, %1, off"
                 :: "v"(lo), "v"(g) : "memory");
  };

  float v = 0.0f;   // pair-equal path metric, lane p holds v[p]
  issue(0);

  for (int k = 0; k < NTILES; ++k) {
    if (k + 1 < NTILES) {
      issue(k + 1);
      asm volatile("s_wait_asynccnt 1" ::: "memory");   // tile k complete
    } else {
      asm volatile("s_wait_asynccnt 0" ::: "memory");
    }
    const float* ty = &lds_y[wv][hi][k & 1][0];

    if (k == 0) {
      // ---- warm-up: ACS only, no bit emission ----
#pragma unroll 4
      for (int i = 0; i < TILE; ++i) {
        float yt = ty[i];
        float sp = v + fabsf(yt - isi_p);
        float a0 = __int_as_float(
            __builtin_amdgcn_ds_bpermute(i0, __float_as_int(sp)));
        float a1 = __int_as_float(
            __builtin_amdgcn_ds_bpermute(i1, __float_as_int(sp)));
        v = fminf(a0, a1);
      }
      // chunk 0 starts exactly like the reference: in_prob = zeros at t=0
      v = (chunk == 0) ? 0.0f : v;
    } else {
      const int tb = t0 + (k - 1) * TILE;
#pragma unroll 2
      for (int i = 0; i < TILE; ++i) {
        // bit = argmin(in_prob) & 1, tie-break lowest index (BEFORE update)
        float m = v;
        SWZ_MIN(m, 0x041F);   // xor 1
        SWZ_MIN(m, 0x081F);   // xor 2
        SWZ_MIN(m, 0x101F);   // xor 4
        SWZ_MIN(m, 0x201F);   // xor 8
        unsigned bal = (unsigned)__ballot(v == m);
        unsigned hm  = (bal >> halfBase) & 0xFFFFu;
        int idx = __ffs((int)hm) - 1;
        if (l15 == 0) out[tb + i] = (float)(idx & 1);

        // add-compare-select
        float yt = ty[i];
        float sp = v + fabsf(yt - isi_p);
        float a0 = __int_as_float(
            __builtin_amdgcn_ds_bpermute(i0, __float_as_int(sp)));
        float a1 = __int_as_float(
            __builtin_amdgcn_ds_bpermute(i1, __float_as_int(sp)));
        v = fminf(a0, a1);
      }
    }
  }
}

extern "C" void kernel_launch(void* const* d_in, const int* in_sizes, int n_in,
                              void* d_out, int out_size, void* d_ws,
                              size_t ws_size, hipStream_t stream) {
  const float* y     = (const float*)d_in[0];
  const float* gamma = (const float*)d_in[1];
  float*       out   = (float*)d_out;
  (void)in_sizes; (void)n_in; (void)out_size; (void)d_ws; (void)ws_size;

  va_viterbi_kernel<<<dim3(NBLOCKS), dim3(256), 0, stream>>>(y, gamma, out);
}
